// UltraFastLRT_38345468019303
// MI455X (gfx1250) — compile-verified, matrix operand order checked by reading, not run on hardware
//
#include <hip/hip_runtime.h>
#include <hip/hip_bf16.h>

#define H     1024
#define HEADS 16
#define HD    64
#define FF    4096
#define T     65
#define NPOL  4096
#define MAXSTEP 64

typedef __attribute__((ext_vector_type(16))) __bf16 v16bf;
typedef __attribute__((ext_vector_type(8)))  float  v8f;
typedef __attribute__((ext_vector_type(4)))  unsigned int v4u;

struct State { int active; int adaptive_max; };

__device__ __forceinline__ unsigned short f2bf(float f) {
  unsigned u = __float_as_uint(f);
  u += 0x7FFFu + ((u >> 16) & 1u);          // round-to-nearest-even
  return (unsigned short)(u >> 16);
}
__device__ __forceinline__ float bf2f(unsigned short h) {
  return __uint_as_float(((unsigned)h) << 16);
}
__device__ __forceinline__ float sigmoidf(float x) { return 1.f / (1.f + __expf(-x)); }

// ---- CDNA5 async memory->LDS staging (GLOBAL_LOAD_ASYNC_TO_LDS_B128, ASYNCcnt) ----
// dst must be a __shared__ buffer (generic pointer low 32 bits = LDS byte offset),
// nbytes must be a multiple of 16*nthreads.
__device__ __forceinline__ void async_stage(float* dst, const float* __restrict__ src,
                                            int nbytes, int tid, int nthreads) {
  unsigned lbase = (unsigned)(uintptr_t)dst;
  const char* gbase = (const char*)src;
  for (int off = tid * 16; off < nbytes; off += nthreads * 16) {
    unsigned laddr = lbase + (unsigned)off;
    const char* gaddr = gbase + off;
    asm volatile("global_load_async_to_lds_b128 %0, %1, off"
                 :: "v"(laddr), "v"(gaddr) : "memory");
  }
  asm volatile("s_wait_asynccnt 0x0" ::: "memory");
}

// ---- wave32 block reductions (broadcast result to all threads) ----
__device__ __forceinline__ float block_sum(float v, float* rbuf) {
  __syncthreads();
  for (int o = 16; o > 0; o >>= 1) v += __shfl_xor(v, o, 32);
  int wid = threadIdx.x >> 5, lane = threadIdx.x & 31;
  if (lane == 0) rbuf[wid] = v;
  __syncthreads();
  int nw = (blockDim.x + 31) >> 5;
  if (wid == 0) {
    float x = (lane < nw) ? rbuf[lane] : 0.f;
    for (int o = 16; o > 0; o >>= 1) x += __shfl_xor(x, o, 32);
    if (lane == 0) rbuf[0] = x;
  }
  __syncthreads();
  return rbuf[0];
}
__device__ __forceinline__ float block_max(float v, float* rbuf) {
  __syncthreads();
  for (int o = 16; o > 0; o >>= 1) v = fmaxf(v, __shfl_xor(v, o, 32));
  int wid = threadIdx.x >> 5, lane = threadIdx.x & 31;
  if (lane == 0) rbuf[wid] = v;
  __syncthreads();
  int nw = (blockDim.x + 31) >> 5;
  if (wid == 0) {
    float x = (lane < nw) ? rbuf[lane] : -INFINITY;
    for (int o = 16; o > 0; o >>= 1) x = fmaxf(x, __shfl_xor(x, o, 32));
    if (lane == 0) rbuf[0] = x;
  }
  __syncthreads();
  return rbuf[0];
}

// ---- fp32 -> bf16 conversion (row-major copy) ----
__global__ void k_cvt_bf16(const float* __restrict__ s, unsigned short* __restrict__ d, int n) {
  for (int i = blockIdx.x * blockDim.x + threadIdx.x; i < n; i += gridDim.x * blockDim.x)
    d[i] = f2bf(s[i]);
}

// ---- fp32 -> bf16 transposed copy: d[n*rows + k] = bf16(s[k*cols + n]) ----
__global__ void k_cvt_bf16_t(const float* __restrict__ s, unsigned short* __restrict__ d,
                             int rows, int cols) {
  __shared__ float tile[32][33];
  int k0 = blockIdx.y * 32, n0 = blockIdx.x * 32;
  int tx = threadIdx.x, ty = threadIdx.y;          // block (32, 8)
  for (int i = ty; i < 32; i += 8)
    tile[i][tx] = s[(size_t)(k0 + i) * cols + (n0 + tx)];
  __syncthreads();
  for (int i = ty; i < 32; i += 8)
    d[(size_t)(n0 + i) * rows + (k0 + tx)] = f2bf(tile[tx][i]);
}

// ---- board encoder stage 1: pe = emb[pieces]; pos = pe @ Wpe + bpe ----
__global__ void k_encode1(const int* __restrict__ pieces, const float* __restrict__ emb,
                          const float* __restrict__ Wpe, const float* __restrict__ bpe,
                          float* __restrict__ pe, float* __restrict__ pos) {
  int row = blockIdx.x;       // 64
  int t   = threadIdx.x;      // 256
  __shared__ float sp[256];
  int pc = pieces[row];
  float v = emb[pc * 256 + t];
  sp[t] = v;
  pe[row * 256 + t] = v;
  __syncthreads();
  float acc = bpe[t];
  for (int c = 0; c < 256; ++c) acc = fmaf(sp[c], Wpe[c * 256 + t], acc);
  pos[row * 256 + t] = acc;
}

// ---- board encoder stage 2: concat @ Wcp + bcp -> LN0 -> bf16 board, complexity partial ----
__global__ void k_encode2(const float* __restrict__ Wcp, const float* __restrict__ bcp,
                          const float* __restrict__ ln0s, const float* __restrict__ ln0b,
                          const float* __restrict__ w_c,
                          const float* __restrict__ pe, const float* __restrict__ pos,
                          unsigned short* __restrict__ boardB, float* __restrict__ rowc) {
  int row = blockIdx.x;       // 64
  int t   = threadIdx.x;      // 256
  __shared__ float cat[512];
  __shared__ float out[1024];
  __shared__ float rbuf[33];
  cat[t]       = pe[row * 256 + t];
  cat[256 + t] = pos[row * 256 + t];
  __syncthreads();
  for (int jj = 0; jj < 4; ++jj) {
    int j = t + jj * 256;
    float acc = bcp[j];
    for (int c = 0; c < 512; ++c) acc = fmaf(cat[c], Wcp[c * H + j], acc);
    out[j] = acc;
  }
  __syncthreads();
  float s = out[t] + out[t + 256] + out[t + 512] + out[t + 768];
  float mu = block_sum(s, rbuf) * (1.f / 1024.f);
  float d0 = out[t] - mu, d1 = out[t + 256] - mu, d2 = out[t + 512] - mu, d3 = out[t + 768] - mu;
  float var = block_sum(d0 * d0 + d1 * d1 + d2 * d2 + d3 * d3, rbuf) * (1.f / 1024.f);
  float inv = rsqrtf(var + 1e-6f);
  float cc = 0.f;
  for (int jj = 0; jj < 4; ++jj) {
    int j = t + jj * 256;
    float b = (out[j] - mu) * inv * ln0s[j] + ln0b[j];
    boardB[row * H + j] = f2bf(b);
    cc = fmaf(b, w_c[j], cc);
  }
  float tot = block_sum(cc, rbuf);
  if (t == 0) rowc[row] = tot;
}

// ---- complexity gate + state/token init ----
__global__ void k_gates(const float* __restrict__ rowc, const float* __restrict__ b_c,
                        const int* __restrict__ max_steps, const float* __restrict__ init_token,
                        float* __restrict__ token, State* st) {
  __shared__ float rbuf[33];
  int t = threadIdx.x;        // 1024
  token[t] = init_token[t];
  float v = (t < 64) ? rowc[t] : 0.f;
  float total = block_sum(v, rbuf);
  if (t == 0) {
    float sig = sigmoidf(total * (1.f / 64.f) + b_c[0]);
    int ms = max_steps[0];
    int am = (int)floorf(4.0f + sig * (float)(ms - 4));
    st->adaptive_max = am;
    st->active = (am > 0) ? 1 : 0;
  }
}

// ---- WMMA precompute: Kc/Vc[0..63] = board @ {Wk,Wv} + bias ----
// A: boardB row-major; B: WkT/WvT stored N-major so each lane's 16 bf16 values are
// two contiguous 16-byte runs (per ISA 16-bit A/B VGPR layout) -> b128 loads.
__global__ void k_kv_wmma(const unsigned short* __restrict__ boardB,
                          const unsigned short* __restrict__ WkT,
                          const unsigned short* __restrict__ WvT,
                          const float* __restrict__ bk, const float* __restrict__ bv,
                          float* __restrict__ Kc, float* __restrict__ Vc) {
  int tile = blockIdx.x;            // 256 tiles (4 M-tiles x 64 N-tiles)
  int wave = threadIdx.x >> 5;      // blockDim = 64 -> wave 0: K, wave 1: V
  int lane = threadIdx.x & 31;
  const unsigned short* WT = wave ? WvT : WkT;
  const float* bias = wave ? bv : bk;
  float* C = wave ? Vc : Kc;
  int m0 = (tile >> 6) << 4;
  int n0 = (tile & 63) << 4;
  int l15 = lane & 15;
  int kh8 = (lane >> 4) << 3;       // K-half select (0 or 8) per 16-bit operand layout
  const unsigned short* arow = boardB + (size_t)(m0 + l15) * H;
  const unsigned short* brow = WT + (size_t)(n0 + l15) * H;
  union AB { v16bf v; v4u q[2]; };
  AB a0, b0, a1, b1;
  v8f c = {};
  for (int kc = 0; kc < H; kc += 64) {
    a0.q[0] = *(const v4u*)(arow + kc + kh8);
    a0.q[1] = *(const v4u*)(arow + kc + 16 + kh8);
    b0.q[0] = *(const v4u*)(brow + kc + kh8);
    b0.q[1] = *(const v4u*)(brow + kc + 16 + kh8);
    a1.q[0] = *(const v4u*)(arow + kc + 32 + kh8);
    a1.q[1] = *(const v4u*)(arow + kc + 48 + kh8);
    b1.q[0] = *(const v4u*)(brow + kc + 32 + kh8);
    b1.q[1] = *(const v4u*)(brow + kc + 48 + kh8);
    c = __builtin_amdgcn_wmma_f32_16x16x32_bf16(false, a0.v, false, b0.v, (short)0, c, false, false);
    c = __builtin_amdgcn_wmma_f32_16x16x32_bf16(false, a1.v, false, b1.v, (short)0, c, false, false);
  }
  int n = n0 + l15;
  int mhi = (lane >> 4) * 8;
#pragma unroll
  for (int r = 0; r < 8; ++r) {
    int m = m0 + r + mhi;
    C[(size_t)m * H + n] = c[r] + bias[n];
  }
}

// ---- per-step: token Q/K/V GEMVs (bf16 weights, coalesced along output dim) ----
__global__ void k_qkv(const State* st, const float* __restrict__ token,
                      const unsigned short* __restrict__ WqB, const unsigned short* __restrict__ WkB,
                      const unsigned short* __restrict__ WvB,
                      const float* __restrict__ bq, const float* __restrict__ bk,
                      const float* __restrict__ bv,
                      float* __restrict__ qt, float* __restrict__ Kc, float* __restrict__ Vc) {
  if (!st->active) return;
  __shared__ __align__(16) float tk[H];
  int t = threadIdx.x;              // 256, grid 12
  async_stage(tk, token, H * 4, t, 256);
  __syncthreads();
  int gid = blockIdx.x * 256 + t;
  int which = gid >> 10;
  int j = gid & (H - 1);
  const unsigned short* W = (which == 0) ? WqB : ((which == 1) ? WkB : WvB);
  float acc = (which == 0) ? bq[j] : ((which == 1) ? bk[j] : bv[j]);
  for (int k = 0; k < H; ++k) {
    if ((k & 127) == 0) __builtin_prefetch(&W[(size_t)(k + 256) * H + j], 0, 0);
    acc = fmaf(tk[k], bf2f(W[(size_t)k * H + j]), acc);
  }
  if (which == 0) qt[j] = acc;
  else if (which == 1) Kc[64 * H + j] = acc;
  else Vc[64 * H + j] = acc;
}

// ---- per-step: attention for the token row over 65 keys ----
__global__ void k_attn(const State* st, const float* __restrict__ qt,
                       const float* __restrict__ Kc, const float* __restrict__ Vc,
                       float* __restrict__ ao) {
  if (!st->active) return;
  int h = blockIdx.x;               // 16 heads
  int t = threadIdx.x;              // 64
  __shared__ float sc[T];
  __shared__ float q[HD];
  q[t] = qt[h * HD + t];
  __syncthreads();
  for (int kk = t; kk < T; kk += 64) {
    const float* kr = Kc + (size_t)kk * H + h * HD;
    float s = 0.f;
    for (int d = 0; d < HD; ++d) s = fmaf(q[d], kr[d], s);
    sc[kk] = s * 0.125f;            // 1/sqrt(64)
  }
  __syncthreads();
  if (t == 0) {
    float mx = sc[0];
    for (int i = 1; i < T; ++i) mx = fmaxf(mx, sc[i]);
    float ssum = 0.f;
    for (int i = 0; i < T; ++i) { float e = __expf(sc[i] - mx); sc[i] = e; ssum += e; }
    float inv = 1.f / ssum;
    for (int i = 0; i < T; ++i) sc[i] *= inv;
  }
  __syncthreads();
  float acc = 0.f;
  for (int kk = 0; kk < T; ++kk) acc = fmaf(sc[kk], Vc[(size_t)kk * H + h * HD + t], acc);
  ao[h * HD + t] = acc;
}

// ---- per-step: O projection + residual ----
__global__ void k_oproj(const State* st, const float* __restrict__ ao,
                        const unsigned short* __restrict__ WoB, const float* __restrict__ bo,
                        const float* __restrict__ token, float* __restrict__ pre1) {
  if (!st->active) return;
  __shared__ __align__(16) float av[H];
  int t = threadIdx.x;              // 128, grid 8
  async_stage(av, ao, H * 4, t, 128);
  __syncthreads();
  int j = blockIdx.x * 128 + t;
  float acc = bo[j] + token[j];
  for (int k = 0; k < H; ++k) acc = fmaf(av[k], bf2f(WoB[(size_t)k * H + j]), acc);
  pre1[j] = acc;
}

// ---- per-step: LayerNorm (1024-vector, one block) ----
__global__ void k_ln(const State* st, const float* __restrict__ x,
                     const float* __restrict__ s, const float* __restrict__ b,
                     float* __restrict__ y) {
  if (!st->active) return;
  __shared__ float rbuf[33];
  int t = threadIdx.x;              // 1024
  float v = x[t];
  float mu = block_sum(v, rbuf) * (1.f / 1024.f);
  float d = v - mu;
  float var = block_sum(d * d, rbuf) * (1.f / 1024.f);
  y[t] = d * rsqrtf(var + 1e-6f) * s[t] + b[t];
}

// ---- per-step: FFN up (1024 -> 4096) + ReLU ----
__global__ void k_ffn1(const State* st, const float* __restrict__ xmid,
                       const unsigned short* __restrict__ W1B, const float* __restrict__ b1,
                       float* __restrict__ hbuf) {
  if (!st->active) return;
  __shared__ __align__(16) float xv[H];
  int t = threadIdx.x;              // 256, grid 16
  async_stage(xv, xmid, H * 4, t, 256);
  __syncthreads();
  int j = blockIdx.x * 256 + t;
  float acc = b1[j];
  for (int k = 0; k < H; ++k) {
    if ((k & 127) == 0) __builtin_prefetch(&W1B[(size_t)(k + 256) * FF + j], 0, 0);
    acc = fmaf(xv[k], bf2f(W1B[(size_t)k * FF + j]), acc);
  }
  hbuf[j] = fmaxf(acc, 0.f);
}

// ---- per-step: FFN down (4096 -> 1024) + residual ----
__global__ void k_ffn2(const State* st, const float* __restrict__ hbuf,
                       const unsigned short* __restrict__ W2B, const float* __restrict__ b2,
                       const float* __restrict__ xmid, float* __restrict__ pre2) {
  if (!st->active) return;
  __shared__ __align__(16) float hv[FF];
  int t = threadIdx.x;              // 128, grid 8
  async_stage(hv, hbuf, FF * 4, t, 128);
  __syncthreads();
  int j = blockIdx.x * 128 + t;
  float acc = b2[j] + xmid[j];
  for (int k = 0; k < FF; ++k) {
    if ((k & 255) == 0) __builtin_prefetch(&W2B[(size_t)(k + 512) * H + j], 0, 0);
    acc = fmaf(hv[k], bf2f(W2B[(size_t)k * H + j]), acc);
  }
  pre2[j] = acc;
}

// ---- per-step: LN2 + gate + token update + stop decision ----
__global__ void k_update(State* st, const float* __restrict__ pre2,
                         const float* __restrict__ ln2s, const float* __restrict__ ln2b,
                         const float* __restrict__ Wg, const float* __restrict__ bg,
                         const float* __restrict__ Wsp, const float* __restrict__ bs,
                         float* __restrict__ token, int step) {
  if (!st->active) return;
  __shared__ float rbuf[33];
  int t = threadIdx.x;              // 1024
  float v = pre2[t];
  float mu = block_sum(v, rbuf) * (1.f / 1024.f);
  float d = v - mu;
  float var = block_sum(d * d, rbuf) * (1.f / 1024.f);
  float nt = d * rsqrtf(var + 1e-6f) * ln2s[t] + ln2b[t];
  float tok = token[t];
  float gdot = block_sum(tok * Wg[t] + nt * Wg[H + t], rbuf);
  float g = sigmoidf(gdot + bg[0]);
  float ut = g * nt + (1.f - g) * tok;
  float sdot = block_sum(ut * Wsp[t], rbuf);
  token[t] = ut;
  if (t == 0) {
    float sp = sigmoidf(sdot + bs[0]);
    int cont = (sp < 0.95f) ? 1 : 0;
    st->active = (cont && (step + 1 < st->adaptive_max)) ? 1 : 0;
  }
}

// ---- heads ----
__global__ void k_value(const float* __restrict__ token, const float* __restrict__ Wvh,
                        const float* __restrict__ bvh, float* __restrict__ out) {
  __shared__ float rbuf[33];
  int t = threadIdx.x;              // 1024
  float tok = token[t];
  float sdot = block_sum(tok * Wvh[t], rbuf);
  if (t == 0) out[0] = tanhf(sdot + bvh[0]);
  out[1 + NPOL + t] = tok;
}

__global__ void k_plogits(const float* __restrict__ token, const float* __restrict__ Wp,
                          const float* __restrict__ bp, float* __restrict__ logits) {
  __shared__ __align__(16) float xv[H];
  int t = threadIdx.x;              // 256, grid 16
  async_stage(xv, token, H * 4, t, 256);
  __syncthreads();
  int j = blockIdx.x * 256 + t;
  float acc = bp[j];
  for (int k = 0; k < H; ++k) acc = fmaf(xv[k], Wp[(size_t)k * NPOL + j], acc);
  logits[j] = acc;
}

__global__ void k_softmax(const float* __restrict__ logits, float* __restrict__ out) {
  __shared__ float rbuf[33];
  int t = threadIdx.x;              // 1024
  float l0 = logits[t], l1 = logits[t + 1024], l2 = logits[t + 2048], l3 = logits[t + 3072];
  float mx = block_max(fmaxf(fmaxf(l0, l1), fmaxf(l2, l3)), rbuf);
  float e0 = __expf(l0 - mx), e1 = __expf(l1 - mx), e2 = __expf(l2 - mx), e3 = __expf(l3 - mx);
  float inv = 1.f / block_sum(e0 + e1 + e2 + e3, rbuf);
  out[1 + t] = e0 * inv;
  out[1 + t + 1024] = e1 * inv;
  out[1 + t + 2048] = e2 * inv;
  out[1 + t + 3072] = e3 * inv;
}

extern "C" void kernel_launch(void* const* d_in, const int* in_sizes, int n_in,
                              void* d_out, int out_size, void* d_ws, size_t ws_size,
                              hipStream_t stream) {
  (void)in_sizes; (void)n_in; (void)out_size; (void)ws_size;
  const int*   pieces     = (const int*)d_in[0];
  const int*   max_steps  = (const int*)d_in[1];
  const float* emb        = (const float*)d_in[2];
  const float* Wpe        = (const float*)d_in[3];
  const float* bpe        = (const float*)d_in[4];
  const float* Wcp        = (const float*)d_in[5];
  const float* bcp        = (const float*)d_in[6];
  const float* ln0s       = (const float*)d_in[7];
  const float* ln0b       = (const float*)d_in[8];
  const float* Wq         = (const float*)d_in[9];
  const float* bq         = (const float*)d_in[10];
  const float* Wk         = (const float*)d_in[11];
  const float* bk         = (const float*)d_in[12];
  const float* Wv         = (const float*)d_in[13];
  const float* bv         = (const float*)d_in[14];
  const float* Wo         = (const float*)d_in[15];
  const float* bo         = (const float*)d_in[16];
  const float* ln1s       = (const float*)d_in[17];
  const float* ln1b       = (const float*)d_in[18];
  const float* W1         = (const float*)d_in[19];
  const float* b1         = (const float*)d_in[20];
  const float* W2         = (const float*)d_in[21];
  const float* b2         = (const float*)d_in[22];
  const float* ln2s       = (const float*)d_in[23];
  const float* ln2b       = (const float*)d_in[24];
  const float* Wvh        = (const float*)d_in[25];
  const float* bvh        = (const float*)d_in[26];
  const float* Wp         = (const float*)d_in[27];
  const float* bp         = (const float*)d_in[28];
  const float* init_token = (const float*)d_in[29];
  const float* w_c        = (const float*)d_in[30];
  const float* b_c        = (const float*)d_in[31];
  const float* Wg         = (const float*)d_in[32];
  const float* bg         = (const float*)d_in[33];
  const float* Wsp        = (const float*)d_in[34];
  const float* bs         = (const float*)d_in[35];
  float* out = (float*)d_out;

  char* w = (char*)d_ws;
  auto alloc = [&](size_t bytes) -> char* {
    char* p = w; w += (bytes + 255) & ~(size_t)255; return p;
  };
  unsigned short* WqB    = (unsigned short*)alloc((size_t)H * H * 2);
  unsigned short* WkB    = (unsigned short*)alloc((size_t)H * H * 2);
  unsigned short* WvB    = (unsigned short*)alloc((size_t)H * H * 2);
  unsigned short* WoB    = (unsigned short*)alloc((size_t)H * H * 2);
  unsigned short* WkT    = (unsigned short*)alloc((size_t)H * H * 2);  // N-major for WMMA B
  unsigned short* WvT    = (unsigned short*)alloc((size_t)H * H * 2);  // N-major for WMMA B
  unsigned short* W1B    = (unsigned short*)alloc((size_t)H * FF * 2);
  unsigned short* W2B    = (unsigned short*)alloc((size_t)FF * H * 2);
  unsigned short* boardB = (unsigned short*)alloc((size_t)64 * H * 2);
  float* Kc     = (float*)alloc((size_t)T * H * 4);
  float* Vc     = (float*)alloc((size_t)T * H * 4);
  float* pe     = (float*)alloc((size_t)64 * 256 * 4);
  float* pos    = (float*)alloc((size_t)64 * 256 * 4);
  float* token  = (float*)alloc(H * 4);
  float* qt     = (float*)alloc(H * 4);
  float* ao     = (float*)alloc(H * 4);
  float* pre1   = (float*)alloc(H * 4);
  float* xmid   = (float*)alloc(H * 4);
  float* hbuf   = (float*)alloc(FF * 4);
  float* pre2   = (float*)alloc(H * 4);
  float* logits = (float*)alloc(NPOL * 4);
  float* rowc   = (float*)alloc(64 * 4);
  State* st     = (State*)alloc(sizeof(State));

  // weight conversion (row-major for GEMV, transposed for WMMA B operands)
  k_cvt_bf16<<<1024, 256, 0, stream>>>(Wq, WqB, H * H);
  k_cvt_bf16<<<1024, 256, 0, stream>>>(Wk, WkB, H * H);
  k_cvt_bf16<<<1024, 256, 0, stream>>>(Wv, WvB, H * H);
  k_cvt_bf16<<<1024, 256, 0, stream>>>(Wo, WoB, H * H);
  k_cvt_bf16<<<2048, 256, 0, stream>>>(W1, W1B, H * FF);
  k_cvt_bf16<<<2048, 256, 0, stream>>>(W2, W2B, FF * H);
  {
    dim3 g(H / 32, H / 32), b(32, 8);
    k_cvt_bf16_t<<<g, b, 0, stream>>>(Wk, WkT, H, H);
    k_cvt_bf16_t<<<g, b, 0, stream>>>(Wv, WvT, H, H);
  }

  // board encoder + gates
  k_encode1<<<64, 256, 0, stream>>>(pieces, emb, Wpe, bpe, pe, pos);
  k_encode2<<<64, 256, 0, stream>>>(Wcp, bcp, ln0s, ln0b, w_c, pe, pos, boardB, rowc);
  k_gates<<<1, 1024, 0, stream>>>(rowc, b_c, max_steps, init_token, token, st);

  // loop-invariant K/V of board rows via WMMA (256 tiles, 2 waves/block)
  k_kv_wmma<<<256, 64, 0, stream>>>(boardB, WkT, WvT, bk, bv, Kc, Vc);

  // fixed-length step schedule; device-side early exit via st->active
  for (int step = 0; step < MAXSTEP; ++step) {
    k_qkv<<<12, 256, 0, stream>>>(st, token, WqB, WkB, WvB, bq, bk, bv, qt, Kc, Vc);
    k_attn<<<HEADS, 64, 0, stream>>>(st, qt, Kc, Vc, ao);
    k_oproj<<<8, 128, 0, stream>>>(st, ao, WoB, bo, token, pre1);
    k_ln<<<1, 1024, 0, stream>>>(st, pre1, ln1s, ln1b, xmid);
    k_ffn1<<<16, 256, 0, stream>>>(st, xmid, W1B, b1, hbuf);
    k_ffn2<<<8, 128, 0, stream>>>(st, hbuf, W2B, b2, xmid, pre2);
    k_update<<<1, 1024, 0, stream>>>(st, pre2, ln2s, ln2b, Wg, bg, Wsp, bs, token, step);
  }

  // heads
  k_value<<<1, 1024, 0, stream>>>(token, Wvh, bvh, out);
  k_plogits<<<16, 256, 0, stream>>>(token, Wp, bp, logits);
  k_softmax<<<1, 1024, 0, stream>>>(logits, out);
}
